// OrbitLSTM_67740224192938
// MI455X (gfx1250) — matrix operand order repaired and checked by reading, not compiled, not run on hardware
//
#include <hip/hip_runtime.h>

// ---------------------------------------------------------------------------
// OrbitLSTM for MI455X (gfx1250, wave32, WMMA 16x16x32 f16->f32)
// Persistent batch-tile-per-wave recurrent kernels; all weights fp16,
// B-operand read row-major, A-operand relayout via per-wave LDS.
// ---------------------------------------------------------------------------

typedef _Float16 v16h __attribute__((ext_vector_type(16)));
typedef _Float16 v8h  __attribute__((ext_vector_type(8)));
typedef float    v8f  __attribute__((ext_vector_type(8)));

constexpr int Bq = 512, Tq = 720, INq = 6, Hq = 128, Gq = 512; // G = 4H
constexpr int OUTq = 3, HORq = 360;

union V16H { v16h v; v8h h[2]; };

// B-operand (32x16, f16): b[i] <- p[i], contiguous 16 halves (k = kbase+16*half+i)
__device__ __forceinline__ v16h load_b16(const _Float16* __restrict__ p) {
  V16H u; u.h[0] = *(const v8h*)p; u.h[1] = *(const v8h*)(p + 8); return u.v;
}
// A-operand (16x32, f16): a[i] <- p[i], contiguous 16 halves (k = kbase+16*half+i)
__device__ __forceinline__ v16h load_a16(const _Float16* __restrict__ p) {
  V16H u; u.h[0] = *(const v8h*)p; u.h[1] = *(const v8h*)(p + 8); return u.v;
}

__device__ __forceinline__ v8f wmma16(v16h a, v16h b, v8f c) {
  return __builtin_amdgcn_wmma_f32_16x16x32_f16(false, a, false, b, (short)0, c,
                                                false, false);
}

__device__ __forceinline__ float sigm(float x) { return 1.f / (1.f + __expf(-x)); }
__device__ __forceinline__ float tanh_fast(float x) {
  float xc = fminf(fmaxf(x, -15.f), 15.f);
  float e = __expf(2.f * xc);
  return (e - 1.f) / (e + 1.f);
}

// ---------------------------------------------------------------------------
// Weight prep: fp32 -> fp16, row-major [rows][dk], zero-pad K from sk to dk
// ---------------------------------------------------------------------------
__global__ void cvt_pad_f16(const float* __restrict__ s, _Float16* __restrict__ d,
                            int rows, int sk, int dk) {
  int i = blockIdx.x * blockDim.x + threadIdx.x;
  if (i >= rows * dk) return;
  int r = i / dk, k = i % dk;
  d[i] = (_Float16)(k < sk ? s[r * sk + k] : 0.f);
}

__global__ void bias_sum(const float* __restrict__ a, const float* __restrict__ b,
                         float* __restrict__ o, int n) {
  int i = blockIdx.x * blockDim.x + threadIdx.x;
  if (i < n) o[i] = a[i] + b[i];
}

// ---------------------------------------------------------------------------
// Encoder layer 0: x (B,T,6) -> y0 half (B,T,256), finals hfin/cfin (B,128)
// Block = 128 threads (4 waves), each wave owns a 16-row batch tile.
// ---------------------------------------------------------------------------
__launch_bounds__(128, 1)
__global__ void enc0_kernel(const float* __restrict__ x,
                            const _Float16* __restrict__ Wih,  // 512 x 32 (K padded 6->32)
                            const _Float16* __restrict__ Whh,  // 512 x 128
                            const float* __restrict__ bias,    // 512  (bih+bhh)
                            _Float16* __restrict__ y0,         // B,T,256
                            float* __restrict__ hfin, float* __restrict__ cfin,
                            int reverse, int dirofs) {
  __shared__ __align__(16) _Float16 hbuf_s[4][16 * Hq];
  __shared__ __align__(16) float    cbuf_s[4][16 * Hq];
  __shared__ float bias_s[Gq];
  const int wave = threadIdx.x >> 5, lane = threadIdx.x & 31;
  const int half = lane >> 4, nn = lane & 15;
  const int b0 = blockIdx.x * 64 + wave * 16;
  _Float16* hbuf = hbuf_s[wave];
  float*    cbuf = cbuf_s[wave];
  for (int i = threadIdx.x; i < Gq; i += blockDim.x) bias_s[i] = bias[i];
  for (int i = lane; i < 16 * Hq; i += 32) { hbuf[i] = (_Float16)0.f; cbuf[i] = 0.f; }
  __syncthreads();

  for (int s = 0; s < Tq; ++s) {
    const int t = reverse ? (Tq - 1 - s) : s;
    // A0: x_t padded to K=32 (only half==0 lanes, k<6 non-zero)
    V16H a0;
#pragma unroll
    for (int i = 0; i < 8; i++) { a0.h[0][i] = (_Float16)0.f; a0.h[1][i] = (_Float16)0.f; }
    if (half == 0) {
      const float* xp = x + ((size_t)(b0 + nn) * Tq + t) * INq;
#pragma unroll
      for (int i = 0; i < INq; i++) a0.h[0][i] = (_Float16)xp[i];
    }
    // A: h_{s-1} from LDS (K=128, 4 chunks)
    v16h hA[4];
#pragma unroll
    for (int kc = 0; kc < 4; kc++)
      hA[kc] = load_a16(hbuf + nn * Hq + kc * 32 + 16 * half);

    for (int j = 0; j < 8; j++) {
      v8f acc[4];
#pragma unroll
      for (int g = 0; g < 4; g++) {
        const int n = g * Hq + j * 16 + nn;
        const float bv = bias_s[n];
        v8f c;
#pragma unroll
        for (int i = 0; i < 8; i++) c[i] = bv;
        c = wmma16(a0.v, load_b16(Wih + (size_t)n * 32 + 16 * half), c);
#pragma unroll
        for (int kc = 0; kc < 4; kc++)
          c = wmma16(hA[kc], load_b16(Whh + (size_t)n * Hq + kc * 32 + 16 * half), c);
        acc[g] = c;
      }
      const int col = j * 16 + nn;
#pragma unroll
      for (int r = 0; r < 8; r++) {
        const int m = r + 8 * half;
        float cv = cbuf[m * Hq + col];
        cv = sigm(acc[1][r]) * cv + sigm(acc[0][r]) * tanh_fast(acc[2][r]);
        const float hv = sigm(acc[3][r]) * tanh_fast(cv);
        cbuf[m * Hq + col] = cv;
        hbuf[m * Hq + col] = (_Float16)hv;
        y0[((size_t)(b0 + m) * Tq + t) * (2 * Hq) + dirofs + col] = (_Float16)hv;
      }
    }
  }
#pragma unroll
  for (int j = 0; j < 8; j++) {
    const int col = j * 16 + nn;
#pragma unroll
    for (int r = 0; r < 8; r++) {
      const int m = r + 8 * half;
      hfin[(size_t)(b0 + m) * Hq + col] = (float)hbuf[m * Hq + col];
      cfin[(size_t)(b0 + m) * Hq + col] = cbuf[m * Hq + col];
    }
  }
}

// ---------------------------------------------------------------------------
// Encoder layer 1: y0 (B,T,256) fp16 -> finals only
// ---------------------------------------------------------------------------
__launch_bounds__(128, 1)
__global__ void enc1_kernel(const _Float16* __restrict__ y0,
                            const _Float16* __restrict__ Wih,  // 512 x 256
                            const _Float16* __restrict__ Whh,  // 512 x 128
                            const float* __restrict__ bias,
                            float* __restrict__ hfin, float* __restrict__ cfin,
                            int reverse) {
  __shared__ __align__(16) _Float16 hbuf_s[4][16 * Hq];
  __shared__ __align__(16) float    cbuf_s[4][16 * Hq];
  __shared__ float bias_s[Gq];
  const int wave = threadIdx.x >> 5, lane = threadIdx.x & 31;
  const int half = lane >> 4, nn = lane & 15;
  const int b0 = blockIdx.x * 64 + wave * 16;
  _Float16* hbuf = hbuf_s[wave];
  float*    cbuf = cbuf_s[wave];
  for (int i = threadIdx.x; i < Gq; i += blockDim.x) bias_s[i] = bias[i];
  for (int i = lane; i < 16 * Hq; i += 32) { hbuf[i] = (_Float16)0.f; cbuf[i] = 0.f; }
  __syncthreads();

  for (int s = 0; s < Tq; ++s) {
    const int t = reverse ? (Tq - 1 - s) : s;
    const _Float16* yrow = y0 + ((size_t)(b0 + nn) * Tq + t) * (2 * Hq);
    if (s + 1 < Tq) {
      const int tn = reverse ? (Tq - 2 - s) : (s + 1);
      __builtin_prefetch(y0 + ((size_t)(b0 + nn) * Tq + tn) * (2 * Hq), 0, 1);
    }
    v16h yA[8];
#pragma unroll
    for (int kc = 0; kc < 8; kc++)
      yA[kc] = load_a16(yrow + kc * 32 + 16 * half);
    v16h hA[4];
#pragma unroll
    for (int kc = 0; kc < 4; kc++)
      hA[kc] = load_a16(hbuf + nn * Hq + kc * 32 + 16 * half);

    for (int j = 0; j < 8; j++) {
      v8f acc[4];
#pragma unroll
      for (int g = 0; g < 4; g++) {
        const int n = g * Hq + j * 16 + nn;
        const float bv = bias_s[n];
        v8f c;
#pragma unroll
        for (int i = 0; i < 8; i++) c[i] = bv;
#pragma unroll
        for (int kc = 0; kc < 8; kc++)
          c = wmma16(yA[kc], load_b16(Wih + (size_t)n * 256 + kc * 32 + 16 * half), c);
#pragma unroll
        for (int kc = 0; kc < 4; kc++)
          c = wmma16(hA[kc], load_b16(Whh + (size_t)n * Hq + kc * 32 + 16 * half), c);
        acc[g] = c;
      }
      const int col = j * 16 + nn;
#pragma unroll
      for (int r = 0; r < 8; r++) {
        const int m = r + 8 * half;
        float cv = cbuf[m * Hq + col];
        cv = sigm(acc[1][r]) * cv + sigm(acc[0][r]) * tanh_fast(acc[2][r]);
        const float hv = sigm(acc[3][r]) * tanh_fast(cv);
        cbuf[m * Hq + col] = cv;
        hbuf[m * Hq + col] = (_Float16)hv;
      }
    }
  }
#pragma unroll
  for (int j = 0; j < 8; j++) {
    const int col = j * 16 + nn;
#pragma unroll
    for (int r = 0; r < 8; r++) {
      const int m = r + 8 * half;
      hfin[(size_t)(b0 + m) * Hq + col] = (float)hbuf[m * Hq + col];
      cfin[(size_t)(b0 + m) * Hq + col] = cbuf[m * Hq + col];
    }
  }
}

// ---------------------------------------------------------------------------
// Bridge: out[l,b,:] = concat(hf[l][b], hb[l][b]) @ W^T + bias   (W: 128x256)
// ---------------------------------------------------------------------------
__global__ void bridge_kernel(const float* __restrict__ hf0, const float* __restrict__ hb0,
                              const float* __restrict__ hf1, const float* __restrict__ hb1,
                              const float* __restrict__ W, const float* __restrict__ b,
                              float* __restrict__ outp) {
  int idx = blockIdx.x * blockDim.x + threadIdx.x;
  if (idx >= 2 * Bq * Hq) return;
  const int l = idx / (Bq * Hq), rem = idx % (Bq * Hq), bi = rem / Hq, o = rem % Hq;
  const float* hf = l ? hf1 : hf0;
  const float* hb = l ? hb1 : hb0;
  float acc = b[o];
  for (int k = 0; k < Hq; k++) acc += hf[(size_t)bi * Hq + k] * W[o * 2 * Hq + k];
  for (int k = 0; k < Hq; k++) acc += hb[(size_t)bi * Hq + k] * W[o * 2 * Hq + Hq + k];
  outp[idx] = acc;
}

// ---------------------------------------------------------------------------
// Decoder: 2 LSTM cells, 360 autoregressive steps, output projection H->3.
// Block = 64 threads (2 waves), each wave owns a 16-row batch tile.
// ---------------------------------------------------------------------------
__launch_bounds__(64, 1)
__global__ void dec_kernel(const _Float16* __restrict__ W0ih,  // 512 x 32 (K padded 3->32)
                           const _Float16* __restrict__ W0hh,  // 512 x 128
                           const _Float16* __restrict__ W1ih,  // 512 x 128
                           const _Float16* __restrict__ W1hh,  // 512 x 128
                           const float* __restrict__ bias0, const float* __restrict__ bias1,
                           const float* __restrict__ hdec, const float* __restrict__ cdec,
                           const float* __restrict__ outW, const float* __restrict__ outB,
                           const float* __restrict__ start,
                           float* __restrict__ out) {
  __shared__ __align__(16) _Float16 h0s[2][16 * Hq], h1s[2][16 * Hq];
  __shared__ __align__(16) float    c0s[2][16 * Hq], c1s[2][16 * Hq];
  __shared__ float preds[2][16 * 4];
  __shared__ float b0_s[Gq], b1_s[Gq], ow_s[OUTq * Hq], ob_s[OUTq], st_s[OUTq];
  const int wave = threadIdx.x >> 5, lane = threadIdx.x & 31;
  const int half = lane >> 4, nn = lane & 15;
  const int b0 = blockIdx.x * 32 + wave * 16;
  _Float16* h0 = h0s[wave]; _Float16* h1 = h1s[wave];
  float* c0 = c0s[wave];    float* c1 = c1s[wave];
  float* pr = preds[wave];
  for (int i = threadIdx.x; i < Gq; i += blockDim.x) { b0_s[i] = bias0[i]; b1_s[i] = bias1[i]; }
  for (int i = threadIdx.x; i < OUTq * Hq; i += blockDim.x) ow_s[i] = outW[i];
  if (threadIdx.x < OUTq) { ob_s[threadIdx.x] = outB[threadIdx.x]; st_s[threadIdx.x] = start[threadIdx.x]; }
  for (int i = lane; i < 16 * Hq; i += 32) {
    const int m = i / Hq, k = i % Hq;
    h0[i] = (_Float16)hdec[(size_t)(b0 + m) * Hq + k];
    c0[i] = cdec[(size_t)(b0 + m) * Hq + k];
    h1[i] = (_Float16)hdec[(size_t)(Bq + b0 + m) * Hq + k];
    c1[i] = cdec[(size_t)(Bq + b0 + m) * Hq + k];
  }
  __syncthreads();
  if (lane < 16) {
#pragma unroll
    for (int o = 0; o < OUTq; o++) pr[lane * 4 + o] = st_s[o];
  }

  for (int s = 0; s < HORq; s++) {
    // ---- cell 0: xin = prev pred (padded K 3->32), recurrent h0 ----
    V16H a0;
#pragma unroll
    for (int i = 0; i < 8; i++) { a0.h[0][i] = (_Float16)0.f; a0.h[1][i] = (_Float16)0.f; }
    if (half == 0) {
#pragma unroll
      for (int i = 0; i < OUTq; i++) a0.h[0][i] = (_Float16)pr[nn * 4 + i];
    }
    v16h hA[4];
#pragma unroll
    for (int kc = 0; kc < 4; kc++)
      hA[kc] = load_a16(h0 + nn * Hq + kc * 32 + 16 * half);

    for (int j = 0; j < 8; j++) {
      v8f acc[4];
#pragma unroll
      for (int g = 0; g < 4; g++) {
        const int n = g * Hq + j * 16 + nn;
        const float bv = b0_s[n];
        v8f c;
#pragma unroll
        for (int i = 0; i < 8; i++) c[i] = bv;
        c = wmma16(a0.v, load_b16(W0ih + (size_t)n * 32 + 16 * half), c);
#pragma unroll
        for (int kc = 0; kc < 4; kc++)
          c = wmma16(hA[kc], load_b16(W0hh + (size_t)n * Hq + kc * 32 + 16 * half), c);
        acc[g] = c;
      }
      const int col = j * 16 + nn;
#pragma unroll
      for (int r = 0; r < 8; r++) {
        const int m = r + 8 * half;
        float cv = c0[m * Hq + col];
        cv = sigm(acc[1][r]) * cv + sigm(acc[0][r]) * tanh_fast(acc[2][r]);
        const float hv = sigm(acc[3][r]) * tanh_fast(cv);
        c0[m * Hq + col] = cv;
        h0[m * Hq + col] = (_Float16)hv;
      }
    }

    // ---- cell 1: xin = new h0, recurrent h1 ----
    v16h xA[4], gA[4];
#pragma unroll
    for (int kc = 0; kc < 4; kc++) {
      xA[kc] = load_a16(h0 + nn * Hq + kc * 32 + 16 * half);
      gA[kc] = load_a16(h1 + nn * Hq + kc * 32 + 16 * half);
    }
    for (int j = 0; j < 8; j++) {
      v8f acc[4];
#pragma unroll
      for (int g = 0; g < 4; g++) {
        const int n = g * Hq + j * 16 + nn;
        const float bv = b1_s[n];
        v8f c;
#pragma unroll
        for (int i = 0; i < 8; i++) c[i] = bv;
#pragma unroll
        for (int kc = 0; kc < 4; kc++) {
          c = wmma16(xA[kc], load_b16(W1ih + (size_t)n * Hq + kc * 32 + 16 * half), c);
          c = wmma16(gA[kc], load_b16(W1hh + (size_t)n * Hq + kc * 32 + 16 * half), c);
        }
        acc[g] = c;
      }
      const int col = j * 16 + nn;
#pragma unroll
      for (int r = 0; r < 8; r++) {
        const int m = r + 8 * half;
        float cv = c1[m * Hq + col];
        cv = sigm(acc[1][r]) * cv + sigm(acc[0][r]) * tanh_fast(acc[2][r]);
        const float hv = sigm(acc[3][r]) * tanh_fast(cv);
        c1[m * Hq + col] = cv;
        h1[m * Hq + col] = (_Float16)hv;
      }
    }

    // ---- output projection + feedback ----
    if (lane < 16) {
      const int m = nn;
#pragma unroll
      for (int o = 0; o < OUTq; o++) {
        float d = ob_s[o];
        for (int k = 0; k < Hq; k++) d += (float)h1[m * Hq + k] * ow_s[o * Hq + k];
        out[((size_t)(b0 + m) * HORq + s) * OUTq + o] = d;
        pr[m * 4 + o] = d;
      }
    }
  }
}

// ---------------------------------------------------------------------------
// Host launcher
// ---------------------------------------------------------------------------
extern "C" void kernel_launch(void* const* d_in, const int* in_sizes, int n_in,
                              void* d_out, int out_size, void* d_ws, size_t ws_size,
                              hipStream_t stream) {
  (void)in_sizes; (void)n_in; (void)out_size; (void)ws_size;
  const float* x        = (const float*)d_in[0];
  const float* e0f_Wih  = (const float*)d_in[1];
  const float* e0f_Whh  = (const float*)d_in[2];
  const float* e0f_bih  = (const float*)d_in[3];
  const float* e0f_bhh  = (const float*)d_in[4];
  const float* e0b_Wih  = (const float*)d_in[5];
  const float* e0b_Whh  = (const float*)d_in[6];
  const float* e0b_bih  = (const float*)d_in[7];
  const float* e0b_bhh  = (const float*)d_in[8];
  const float* e1f_Wih  = (const float*)d_in[9];
  const float* e1f_Whh  = (const float*)d_in[10];
  const float* e1f_bih  = (const float*)d_in[11];
  const float* e1f_bhh  = (const float*)d_in[12];
  const float* e1b_Wih  = (const float*)d_in[13];
  const float* e1b_Whh  = (const float*)d_in[14];
  const float* e1b_bih  = (const float*)d_in[15];
  const float* e1b_bhh  = (const float*)d_in[16];
  const float* brh_W    = (const float*)d_in[17];
  const float* brh_b    = (const float*)d_in[18];
  const float* brc_W    = (const float*)d_in[19];
  const float* brc_b    = (const float*)d_in[20];
  const float* d0_Wih   = (const float*)d_in[21];
  const float* d0_Whh   = (const float*)d_in[22];
  const float* d0_bih   = (const float*)d_in[23];
  const float* d0_bhh   = (const float*)d_in[24];
  const float* d1_Wih   = (const float*)d_in[25];
  const float* d1_Whh   = (const float*)d_in[26];
  const float* d1_bih   = (const float*)d_in[27];
  const float* d1_bhh   = (const float*)d_in[28];
  const float* out_W    = (const float*)d_in[29];
  const float* out_b    = (const float*)d_in[30];
  const float* start_tk = (const float*)d_in[31];

  char* ws = (char*)d_ws;
  size_t off = 0;
  auto alloc = [&](size_t bytes) -> char* {
    char* p = ws + off;
    off = (off + bytes + 255) & ~(size_t)255;
    return p;
  };

  _Float16* y0    = (_Float16*)alloc((size_t)Bq * Tq * 2 * Hq * 2);
  _Float16* W0ihF = (_Float16*)alloc((size_t)Gq * 32 * 2);
  _Float16* W0ihB = (_Float16*)alloc((size_t)Gq * 32 * 2);
  _Float16* W0hhF = (_Float16*)alloc((size_t)Gq * Hq * 2);
  _Float16* W0hhB = (_Float16*)alloc((size_t)Gq * Hq * 2);
  _Float16* W1ihF = (_Float16*)alloc((size_t)Gq * 256 * 2);
  _Float16* W1ihB = (_Float16*)alloc((size_t)Gq * 256 * 2);
  _Float16* W1hhF = (_Float16*)alloc((size_t)Gq * Hq * 2);
  _Float16* W1hhB = (_Float16*)alloc((size_t)Gq * Hq * 2);
  _Float16* D0ih  = (_Float16*)alloc((size_t)Gq * 32 * 2);
  _Float16* D0hh  = (_Float16*)alloc((size_t)Gq * Hq * 2);
  _Float16* D1ih  = (_Float16*)alloc((size_t)Gq * Hq * 2);
  _Float16* D1hh  = (_Float16*)alloc((size_t)Gq * Hq * 2);
  float* b0F = (float*)alloc(Gq * 4);
  float* b0B = (float*)alloc(Gq * 4);
  float* b1F = (float*)alloc(Gq * 4);
  float* b1B = (float*)alloc(Gq * 4);
  float* bD0 = (float*)alloc(Gq * 4);
  float* bD1 = (float*)alloc(Gq * 4);
  float* h0F = (float*)alloc((size_t)Bq * Hq * 4);
  float* c0F = (float*)alloc((size_t)Bq * Hq * 4);
  float* h0B = (float*)alloc((size_t)Bq * Hq * 4);
  float* c0B = (float*)alloc((size_t)Bq * Hq * 4);
  float* h1F = (float*)alloc((size_t)Bq * Hq * 4);
  float* c1F = (float*)alloc((size_t)Bq * Hq * 4);
  float* h1B = (float*)alloc((size_t)Bq * Hq * 4);
  float* c1B = (float*)alloc((size_t)Bq * Hq * 4);
  float* hdec = (float*)alloc((size_t)2 * Bq * Hq * 4);
  float* cdec = (float*)alloc((size_t)2 * Bq * Hq * 4);

  dim3 thr(256);
  auto g1 = [](int n) { return dim3((unsigned)((n + 255) / 256)); };

  // ---- weight prep (fp32 -> fp16, pad K) ----
  cvt_pad_f16<<<g1(Gq * 32), thr, 0, stream>>>(e0f_Wih, W0ihF, Gq, INq, 32);
  cvt_pad_f16<<<g1(Gq * 32), thr, 0, stream>>>(e0b_Wih, W0ihB, Gq, INq, 32);
  cvt_pad_f16<<<g1(Gq * Hq), thr, 0, stream>>>(e0f_Whh, W0hhF, Gq, Hq, Hq);
  cvt_pad_f16<<<g1(Gq * Hq), thr, 0, stream>>>(e0b_Whh, W0hhB, Gq, Hq, Hq);
  cvt_pad_f16<<<g1(Gq * 256), thr, 0, stream>>>(e1f_Wih, W1ihF, Gq, 256, 256);
  cvt_pad_f16<<<g1(Gq * 256), thr, 0, stream>>>(e1b_Wih, W1ihB, Gq, 256, 256);
  cvt_pad_f16<<<g1(Gq * Hq), thr, 0, stream>>>(e1f_Whh, W1hhF, Gq, Hq, Hq);
  cvt_pad_f16<<<g1(Gq * Hq), thr, 0, stream>>>(e1b_Whh, W1hhB, Gq, Hq, Hq);
  cvt_pad_f16<<<g1(Gq * 32), thr, 0, stream>>>(d0_Wih, D0ih, Gq, OUTq, 32);
  cvt_pad_f16<<<g1(Gq * Hq), thr, 0, stream>>>(d0_Whh, D0hh, Gq, Hq, Hq);
  cvt_pad_f16<<<g1(Gq * Hq), thr, 0, stream>>>(d1_Wih, D1ih, Gq, Hq, Hq);
  cvt_pad_f16<<<g1(Gq * Hq), thr, 0, stream>>>(d1_Whh, D1hh, Gq, Hq, Hq);
  bias_sum<<<g1(Gq), thr, 0, stream>>>(e0f_bih, e0f_bhh, b0F, Gq);
  bias_sum<<<g1(Gq), thr, 0, stream>>>(e0b_bih, e0b_bhh, b0B, Gq);
  bias_sum<<<g1(Gq), thr, 0, stream>>>(e1f_bih, e1f_bhh, b1F, Gq);
  bias_sum<<<g1(Gq), thr, 0, stream>>>(e1b_bih, e1b_bhh, b1B, Gq);
  bias_sum<<<g1(Gq), thr, 0, stream>>>(d0_bih, d0_bhh, bD0, Gq);
  bias_sum<<<g1(Gq), thr, 0, stream>>>(d1_bih, d1_bhh, bD1, Gq);

  // ---- encoder layer 0 (fwd, bwd) ----
  enc0_kernel<<<dim3(Bq / 64), dim3(128), 0, stream>>>(x, W0ihF, W0hhF, b0F, y0, h0F, c0F, 0, 0);
  enc0_kernel<<<dim3(Bq / 64), dim3(128), 0, stream>>>(x, W0ihB, W0hhB, b0B, y0, h0B, c0B, 1, Hq);

  // ---- encoder layer 1 (fwd, bwd) ----
  enc1_kernel<<<dim3(Bq / 64), dim3(128), 0, stream>>>(y0, W1ihF, W1hhF, b1F, h1F, c1F, 0);
  enc1_kernel<<<dim3(Bq / 64), dim3(128), 0, stream>>>(y0, W1ihB, W1hhB, b1B, h1B, c1B, 1);

  // ---- bridge ----
  bridge_kernel<<<g1(2 * Bq * Hq), thr, 0, stream>>>(h0F, h0B, h1F, h1B, brh_W, brh_b, hdec);
  bridge_kernel<<<g1(2 * Bq * Hq), thr, 0, stream>>>(c0F, c0B, c1F, c1B, brc_W, brc_b, cdec);

  // ---- decoder ----
  dec_kernel<<<dim3(Bq / 32), dim3(64), 0, stream>>>(D0ih, D0hh, D1ih, D1hh, bD0, bD1,
                                                     hdec, cdec, out_W, out_b, start_tk,
                                                     (float*)d_out);
}
